// SAN_CLS_50105088475876
// MI455X (gfx1250) — matrix operand order repaired and checked by reading, not compiled, not run on hardware
//
#include <hip/hip_runtime.h>
#include <math.h>

#define BATCH 8
#define SEQ   128
#define MTOT  1024        // BATCH*SEQ
#define VD    300
#define VP    320         // padded V (multiple of 32)
#define NT_M  64          // MTOT/16
#define NB_N  5           // N blocks of 64 (covers 320 >= 300)
#define NTILES (NT_M*NB_N)

typedef __attribute__((ext_vector_type(16))) __bf16        v16bf;
typedef __attribute__((ext_vector_type(8)))  float         v8f;
typedef __attribute__((ext_vector_type(4)))  unsigned int  v4u;

union FragBF16 { v4u u[2]; v16bf v; };

#if defined(__has_builtin)
#  if __has_builtin(__builtin_amdgcn_tanhf)
#    define FAST_TANH(x) __builtin_amdgcn_tanhf(x)
#  else
#    define FAST_TANH(x) tanhf(x)
#  endif
#else
#  define FAST_TANH(x) tanhf(x)
#endif

// exp(5*t) == exp2(t * 5*log2(e)); exp2f maps to a single v_exp_f32
#define EXP5(t) exp2f((t) * 7.2134752044448170f)

__device__ __forceinline__ unsigned short f32_to_bf16(float f) {
  union { float f; unsigned int u; } x; x.f = f;
  unsigned int r = x.u + 0x7FFFu + ((x.u >> 16) & 1u);   // round-to-nearest-even
  return (unsigned short)(r >> 16);
}

// ---- pack float32 weight [VD,VD] -> transposed padded bf16 [VP(n), VP(k)] ----
__global__ void pack_weight_t(const float* __restrict__ W, unsigned short* __restrict__ WT) {
  int idx = blockIdx.x * blockDim.x + threadIdx.x;
  if (idx >= VP * VP) return;
  int n = idx / VP, k = idx % VP;
  float v = (n < VD && k < VD) ? W[k * VD + n] : 0.f;
  WT[idx] = f32_to_bf16(v);
}

// ---- pack float32 activation [MTOT,VD] -> padded bf16 [MTOT,VP] ----
__global__ void pack_act(const float* __restrict__ X, unsigned short* __restrict__ XP) {
  int idx = blockIdx.x * blockDim.x + threadIdx.x;
  if (idx >= MTOT * VP) return;
  int r = idx / VP, c = idx % VP;
  XP[idx] = f32_to_bf16(c < VD ? X[r * VD + c] : 0.f);
}

// ---- WMMA GEMM, 16Mx64N per wave (1 A-frag reused over 4 B-frags) ----
// C[M,VD] = A_bf16[M,VP] x BT_bf16[VP(n),VP(k)]^T (+bias)(+addC)(elu?)
__global__ void __launch_bounds__(256) gemm_bf16_wmma(
    const unsigned short* __restrict__ A, const unsigned short* __restrict__ BT,
    const float* __restrict__ bias, const float* __restrict__ addC,
    float* __restrict__ C, int elu) {
  int wave = threadIdx.x >> 5;
  int lane = threadIdx.x & 31;
  int tile = blockIdx.x * 8 + wave;
  if (tile >= NTILES) return;               // grid sized exactly; never taken
  int tm = tile / NB_N, tnb = tile - tm * NB_N;
  int m0 = tm << 4, n0 = tnb << 6;          // 64-wide N block
  int half = lane >> 4, rc = lane & 15;
  const unsigned short* Ar  = A  + (size_t)(m0 + rc) * VP;  // row of A
  const unsigned short* Bc0 = BT + (size_t)(n0 + rc) * VP;  // first B column group
  v8f acc[4] = {v8f{}, v8f{}, v8f{}, v8f{}};
#pragma unroll
  for (int k0 = 0; k0 < VP; k0 += 32) {
    FragBF16 a;
    // A 16x32 bf16 layout: lane<16 -> K 0..7 & 16..23 ; lane>=16 -> K 8..15 & 24..31
    a.u[0] = *reinterpret_cast<const v4u*>(Ar + k0 + half * 8);
    a.u[1] = *reinterpret_cast<const v4u*>(Ar + k0 + 16 + half * 8);
#pragma unroll
    for (int s = 0; s < 4; ++s) {
      // B 32x16 bf16 layout: lane<16 -> K 0..7 & 16..23 ; lane>=16 -> K 8..15 & 24..31, N = lane&15
      const unsigned short* Bc = Bc0 + (size_t)(s * 16) * VP;
      FragBF16 b;
      b.u[0] = *reinterpret_cast<const v4u*>(Bc + k0 + half * 8);
      b.u[1] = *reinterpret_cast<const v4u*>(Bc + k0 + 16 + half * 8);
      acc[s] = __builtin_amdgcn_wmma_f32_16x16x32_bf16(false, a.v, false, b.v,
                                                       (short)0, acc[s], false, false);
    }
  }
#pragma unroll
  for (int s = 0; s < 4; ++s) {
    int n = n0 + s * 16 + rc;
    if (n < VD) {
      float bv = bias ? bias[n] : 0.f;
#pragma unroll
      for (int r = 0; r < 8; ++r) {
        int m = m0 + r + half * 8;          // C layout: VGPR r -> M = r + 8*half
        float x = acc[s][r] + bv;
        if (addC) x += addC[(size_t)m * VD + n];
        if (elu) x = x > 0.f ? x : (__expf(x) - 1.f);
        C[(size_t)m * VD + n] = x;
      }
    }
  }
}

// ---- directional multi-dim attention (streaming softmax over j, per feature v) ----
__global__ void __launch_bounds__(VP) attn_dir(
    const float* __restrict__ dep, const float* __restrict__ head,
    const float* __restrict__ rmap, const float* __restrict__ fb,
    const int* __restrict__ mask, float* __restrict__ out, int forward) {
  int bi = blockIdx.x;                      // b*SEQ + i
  int b = bi >> 7, i = bi & 127;
  int v = threadIdx.x;
  if (v >= VD) return;
  float h = head[(size_t)bi * VD + v] + fb[v];
  int j0 = forward ? 0 : (i + 1);
  int j1 = forward ? i : SEQ;
  float num = 0.f, den = 0.f;
  for (int j = j0; j < j1; ++j) {
    if (!mask[b * SEQ + j]) continue;       // uniform branch per block
    float d  = dep[(size_t)(b * SEQ + j) * VD + v];
    float th = FAST_TANH((d + h) * 0.2f);   // logits = 5*tanh(.) in [-5,5]
    float w  = EXP5(th);                    // exp safe unshifted
    num += w * rmap[(size_t)(b * SEQ + j) * VD + v];
    den += w;
  }
  out[(size_t)bi * VD + v] = den > 0.f ? num / den : 0.f;
}

// ---- bias_comb = bfw + bbw + o_bias ----
__global__ void add3(const float* a, const float* b, const float* c, float* o) {
  int i = blockIdx.x * blockDim.x + threadIdx.x;
  if (i < VD) o[i] = a[i] + b[i] + c[i];
}

// ---- fusion gate: x_fus = (sig(g)*x_fw + (1-sig(g))*x_bw) * mask ----
__global__ void fuse_gate(const float* __restrict__ gp, const float* __restrict__ xf,
                          const float* __restrict__ xb, const int* __restrict__ mask,
                          float* __restrict__ xfus) {
  int idx = blockIdx.x * blockDim.x + threadIdx.x;
  if (idx >= MTOT * VD) return;
  int m = idx / VD;
  float g = 1.f / (1.f + __expf(-gp[idx]));
  float r = g * xf[idx] + (1.f - g) * xb[idx];
  xfus[idx] = mask[m] ? r : 0.f;
}

// ---- source2token: softmax over L (masked) then weighted sum of x_fus ----
__global__ void __launch_bounds__(VP) s2t(
    const float* __restrict__ map2, const float* __restrict__ xfus,
    const int* __restrict__ mask, float* __restrict__ xcode) {
  int b = blockIdx.x, v = threadIdx.x;
  if (v >= VD) return;
  float mx = -3.0e38f; int any = 0;
  for (int l = 0; l < SEQ; ++l)
    if (mask[b * SEQ + l]) { float x = map2[(size_t)(b * SEQ + l) * VD + v];
                             mx = fmaxf(mx, x); any = 1; }
  float num = 0.f, den = 0.f;
  if (any)
    for (int l = 0; l < SEQ; ++l)
      if (mask[b * SEQ + l]) {
        float w = __expf(map2[(size_t)(b * SEQ + l) * VD + v] - mx);
        num += w * xfus[(size_t)(b * SEQ + l) * VD + v];
        den += w;
      }
  xcode[b * VD + v] = (any && den > 0.f) ? num / den : 0.f;
}

// ---- classifier head: pre = relu(xcode@Wp + bp); out = pre@Wo ----
__global__ void __launch_bounds__(512) head_mlp(
    const float* __restrict__ xcode, const float* __restrict__ Wp,
    const float* __restrict__ bp, const float* __restrict__ Wo,
    float* __restrict__ out) {
  __shared__ float xs[VD];
  __shared__ float ps[VD];
  int b = blockIdx.x, t = threadIdx.x;
  for (int v = t; v < VD; v += 512) xs[v] = xcode[b * VD + v];
  __syncthreads();
  for (int h = t; h < VD; h += 512) {
    float s = bp[h];
    for (int v = 0; v < VD; ++v) s += xs[v] * Wp[v * VD + h];
    ps[h] = fmaxf(s, 0.f);
  }
  __syncthreads();
  if (t < 3) {
    float s = 0.f;
    for (int h = 0; h < VD; ++h) s += ps[h] * Wo[h * 3 + t];
    out[b * 3 + t] = s;
  }
}

extern "C" void kernel_launch(void* const* d_in, const int* in_sizes, int n_in,
                              void* d_out, int out_size, void* d_ws, size_t ws_size,
                              hipStream_t stream) {
  (void)in_sizes; (void)n_in; (void)out_size; (void)ws_size;
  const float* rep   = (const float*)d_in[0];
  const int*   mask  = (const int*)  d_in[1];
  const float* Wm_f  = (const float*)d_in[2];
  const float* bm_f  = (const float*)d_in[3];
  const float* Wd_f  = (const float*)d_in[4];
  const float* Wh_f  = (const float*)d_in[5];
  const float* fb_f  = (const float*)d_in[6];
  const float* Wm_b  = (const float*)d_in[7];
  const float* bm_b  = (const float*)d_in[8];
  const float* Wd_b  = (const float*)d_in[9];
  const float* Wh_b  = (const float*)d_in[10];
  const float* fb_b  = (const float*)d_in[11];
  const float* Wfw   = (const float*)d_in[12];
  const float* bfw   = (const float*)d_in[13];
  const float* Wbw   = (const float*)d_in[14];
  const float* bbw   = (const float*)d_in[15];
  const float* obias = (const float*)d_in[16];
  const float* W1    = (const float*)d_in[17];
  const float* b1    = (const float*)d_in[18];
  const float* W2    = (const float*)d_in[19];
  const float* b2    = (const float*)d_in[20];
  const float* Wp    = (const float*)d_in[21];
  const float* bp    = (const float*)d_in[22];
  const float* Wo    = (const float*)d_in[23];
  float* out = (float*)d_out;

  // ---- workspace carve-up ----
  char* base = (char*)d_ws;
  size_t off = 0;
  auto alloc = [&](size_t bytes) { void* p = base + off;
                                   off = (off + bytes + 255) & ~(size_t)255; return p; };
  const size_t WT_BYTES  = (size_t)VP * VP * 2;
  const size_t ACT_BYTES = (size_t)MTOT * VP * 2;
  const size_t F32_BYTES = (size_t)MTOT * VD * 4;

  unsigned short* WTm_f = (unsigned short*)alloc(WT_BYTES);
  unsigned short* WTd_f = (unsigned short*)alloc(WT_BYTES);
  unsigned short* WTh_f = (unsigned short*)alloc(WT_BYTES);
  unsigned short* WTm_b = (unsigned short*)alloc(WT_BYTES);
  unsigned short* WTd_b = (unsigned short*)alloc(WT_BYTES);
  unsigned short* WTh_b = (unsigned short*)alloc(WT_BYTES);
  unsigned short* WTfw  = (unsigned short*)alloc(WT_BYTES);
  unsigned short* WTbw  = (unsigned short*)alloc(WT_BYTES);
  unsigned short* WT1   = (unsigned short*)alloc(WT_BYTES);
  unsigned short* WT2   = (unsigned short*)alloc(WT_BYTES);

  unsigned short* repP  = (unsigned short*)alloc(ACT_BYTES);
  unsigned short* rmfP  = (unsigned short*)alloc(ACT_BYTES);
  unsigned short* rmbP  = (unsigned short*)alloc(ACT_BYTES);
  unsigned short* xfwP  = (unsigned short*)alloc(ACT_BYTES);
  unsigned short* xbwP  = (unsigned short*)alloc(ACT_BYTES);
  unsigned short* xfusP = (unsigned short*)alloc(ACT_BYTES);
  unsigned short* m1P   = (unsigned short*)alloc(ACT_BYTES);

  float* rep_map_f = (float*)alloc(F32_BYTES);
  float* rep_map_b = (float*)alloc(F32_BYTES);
  float* dep_f     = (float*)alloc(F32_BYTES);
  float* head_f    = (float*)alloc(F32_BYTES);
  float* dep_b     = (float*)alloc(F32_BYTES);
  float* head_b    = (float*)alloc(F32_BYTES);
  float* x_fw      = (float*)alloc(F32_BYTES);
  float* x_bw      = (float*)alloc(F32_BYTES);
  float* g1        = (float*)alloc(F32_BYTES);
  float* g2        = (float*)alloc(F32_BYTES);
  float* x_fus     = (float*)alloc(F32_BYTES);
  float* map1      = (float*)alloc(F32_BYTES);
  float* map2      = (float*)alloc(F32_BYTES);
  float* bias_comb = (float*)alloc(VD * 4);
  float* x_code    = (float*)alloc(BATCH * VD * 4);

  const int PB = 256;
  dim3 gW((VP * VP + PB - 1) / PB), gA((MTOT * VP + PB - 1) / PB);
  dim3 gG((NTILES + 7) / 8);                // 40 blocks x 8 waves = exactly NTILES
  dim3 gE((MTOT * VD + PB - 1) / PB);

  // 1) pack weights (transposed, padded, bf16)
  pack_weight_t<<<gW, PB, 0, stream>>>(Wm_f, WTm_f);
  pack_weight_t<<<gW, PB, 0, stream>>>(Wd_f, WTd_f);
  pack_weight_t<<<gW, PB, 0, stream>>>(Wh_f, WTh_f);
  pack_weight_t<<<gW, PB, 0, stream>>>(Wm_b, WTm_b);
  pack_weight_t<<<gW, PB, 0, stream>>>(Wd_b, WTd_b);
  pack_weight_t<<<gW, PB, 0, stream>>>(Wh_b, WTh_b);
  pack_weight_t<<<gW, PB, 0, stream>>>(Wfw,  WTfw);
  pack_weight_t<<<gW, PB, 0, stream>>>(Wbw,  WTbw);
  pack_weight_t<<<gW, PB, 0, stream>>>(W1,   WT1);
  pack_weight_t<<<gW, PB, 0, stream>>>(W2,   WT2);
  add3<<<dim3(2), PB, 0, stream>>>(bfw, bbw, obias, bias_comb);

  // 2) rep_map = elu(rep @ Wm + bm) for both directions
  pack_act<<<gA, PB, 0, stream>>>(rep, repP);
  gemm_bf16_wmma<<<gG, 256, 0, stream>>>(repP, WTm_f, bm_f, nullptr, rep_map_f, 1);
  gemm_bf16_wmma<<<gG, 256, 0, stream>>>(repP, WTm_b, bm_b, nullptr, rep_map_b, 1);
  pack_act<<<gA, PB, 0, stream>>>(rep_map_f, rmfP);
  pack_act<<<gA, PB, 0, stream>>>(rep_map_b, rmbP);

  // 3) dep / head projections
  gemm_bf16_wmma<<<gG, 256, 0, stream>>>(rmfP, WTd_f, nullptr, nullptr, dep_f, 0);
  gemm_bf16_wmma<<<gG, 256, 0, stream>>>(rmfP, WTh_f, nullptr, nullptr, head_f, 0);
  gemm_bf16_wmma<<<gG, 256, 0, stream>>>(rmbP, WTd_b, nullptr, nullptr, dep_b, 0);
  gemm_bf16_wmma<<<gG, 256, 0, stream>>>(rmbP, WTh_b, nullptr, nullptr, head_b, 0);

  // 4) directional attention (the transcendental-heavy part)
  attn_dir<<<dim3(MTOT), VP, 0, stream>>>(dep_f, head_f, rep_map_f, fb_f, mask, x_fw, 1);
  attn_dir<<<dim3(MTOT), VP, 0, stream>>>(dep_b, head_b, rep_map_b, fb_b, mask, x_bw, 0);

  // 5) fusion gate: g = x_fw@Wfw + x_bw@Wbw + (bfw+bbw+o_bias)
  pack_act<<<gA, PB, 0, stream>>>(x_fw, xfwP);
  pack_act<<<gA, PB, 0, stream>>>(x_bw, xbwP);
  gemm_bf16_wmma<<<gG, 256, 0, stream>>>(xfwP, WTfw, bias_comb, nullptr, g1, 0);
  gemm_bf16_wmma<<<gG, 256, 0, stream>>>(xbwP, WTbw, nullptr, g1, g2, 0);
  fuse_gate<<<gE, PB, 0, stream>>>(g2, x_fw, x_bw, mask, x_fus);

  // 6) query_encode_md
  pack_act<<<gA, PB, 0, stream>>>(x_fus, xfusP);
  gemm_bf16_wmma<<<gG, 256, 0, stream>>>(xfusP, WT1, b1, nullptr, map1, 1);
  pack_act<<<gA, PB, 0, stream>>>(map1, m1P);
  gemm_bf16_wmma<<<gG, 256, 0, stream>>>(m1P, WT2, b2, nullptr, map2, 0);
  s2t<<<dim3(BATCH), VP, 0, stream>>>(map2, x_fus, mask, x_code);

  // 7) classifier head -> d_out [8,3]
  head_mlp<<<dim3(BATCH), 512, 0, stream>>>(x_code, Wp, bp, Wo, out);
}
